// GraphormerAttentionHead_28484223107288
// MI455X (gfx1250) — matrix operand order repaired and checked by reading, not compiled, not run on hardware
//
#include <hip/hip_runtime.h>
#include <math.h>

#define NTOK 8192
#define DIN  512
#define DH   64

typedef __attribute__((ext_vector_type(16))) __bf16 v16bf;
typedef __attribute__((ext_vector_type(8)))  float  v8f;

struct B32x2 { uint4 a, b; };           // 32 bytes, bit-castable to v16bf
struct H16   { unsigned short h[16]; }; // 32 bytes

// f32 -> bf16 round-to-nearest-even (bit trick; NaN edge ignored)
static __device__ inline unsigned short f2bf(float f) {
  unsigned u = __builtin_bit_cast(unsigned, f);
  u += 0x7FFFu + ((u >> 16) & 1u);
  return (unsigned short)(u >> 16);
}

// B-fragment: h0..h15 = p[0..15] (one contiguous 32B run)
static __device__ inline v16bf ld_b_contig(const unsigned short* p) {
  B32x2 s;
  s.a = *reinterpret_cast<const uint4*>(p);
  s.b = *reinterpret_cast<const uint4*>(p + 8);
  return __builtin_bit_cast(v16bf, s);
}

// A-fragment from a row-major row: h0..15 = row[d0+16*hf ..] (contiguous)
static __device__ inline v16bf ld_a_bf16(const unsigned short* row, int d0, int hf) {
  B32x2 s;
  s.a = *reinterpret_cast<const uint4*>(row + d0 + 16 * hf);
  s.b = *reinterpret_cast<const uint4*>(row + d0 + 16 * hf + 8);
  return __builtin_bit_cast(v16bf, s);
}

static __device__ inline v8f wmma_bf16(v16bf A, v16bf B, v8f C) {
  return __builtin_amdgcn_wmma_f32_16x16x32_bf16(false, A, false, B, (short)0, C,
                                                 false, false);
}

// ---------------------------------------------------------------------------
// Kernel 1: q = 0.125*(x@Wq+bq), k = x@Wk+bk, v = x@Wv+bv  (bf16 outputs)
// q,k row-major [NTOK][64]; v stored transposed vT [64][NTOK] so the flash
// kernel's P@V B-fragments are contiguous 32B runs.
// One wave owns a 16-row tile; W^T is staged in LDS as bf16 (64 KB) per matrix.
// ---------------------------------------------------------------------------
__global__ __launch_bounds__(256) void proj_qkv(
    const float* __restrict__ x,
    const float* __restrict__ Wq, const float* __restrict__ bq,
    const float* __restrict__ Wk, const float* __restrict__ bk,
    const float* __restrict__ Wv, const float* __restrict__ bv,
    unsigned short* __restrict__ qb,
    unsigned short* __restrict__ kb,
    unsigned short* __restrict__ vtb)
{
  __shared__ unsigned short WT[DH * DIN]; // W^T as bf16: WT[n][k]
  const int tid = threadIdx.x;
  const int wv  = tid >> 5;
  const int ln  = tid & 31;
  const int hf  = ln >> 4;
  const int l16 = ln & 15;
  const int row0 = blockIdx.x * 128 + wv * 16;

  const float* Wmat[3] = {Wq, Wk, Wv};
  const float* bvec[3] = {bq, bk, bv};

  for (int m = 0; m < 3; ++m) {
    __syncthreads(); // protect WT reads of previous iteration
    for (int idx = tid; idx < DIN * DH; idx += 256) {
      int kk = idx & (DIN - 1); // fast index -> contiguous LDS writes
      int nn = idx >> 9;
      WT[nn * DIN + kk] = f2bf(Wmat[m][kk * DH + nn]);
    }
    __syncthreads();

    v8f acc[4]; // 4 output col tiles of 16 (DH = 64)
#pragma unroll
    for (int nt = 0; nt < 4; ++nt) {
      float b = bvec[m][nt * 16 + l16]; // bias: C(row,col)=b[col]
#pragma unroll
      for (int r = 0; r < 8; ++r) acc[nt][r] = b;
    }

    const float* xrow = x + (size_t)(row0 + l16) * DIN;
    for (int c = 0; c < DIN; c += 32) {
      // A-fragment of x tile, converted f32 -> bf16
      H16 hh;
      const float* p0 = xrow + c + 16 * hf;
      const float* p1 = xrow + c + 16 * hf + 8;
#pragma unroll
      for (int i = 0; i < 8; ++i) { hh.h[i] = f2bf(p0[i]); hh.h[8 + i] = f2bf(p1[i]); }
      v16bf A = __builtin_bit_cast(v16bf, hh);
#pragma unroll
      for (int nt = 0; nt < 4; ++nt) {
        v16bf B = ld_b_contig(&WT[(nt * 16 + l16) * DIN + c + 16 * hf]);
        acc[nt] = wmma_bf16(A, B, acc[nt]);
      }
    }

    // Epilogue: C/D layout element (r) -> (row0 + r + 8*hf, nt*16 + l16)
    if (m == 0) {
#pragma unroll
      for (int nt = 0; nt < 4; ++nt)
#pragma unroll
        for (int r = 0; r < 8; ++r)
          qb[(size_t)(row0 + r + 8 * hf) * DH + nt * 16 + l16] =
              f2bf(acc[nt][r] * 0.125f); // fold 1/sqrt(64) into q
    } else if (m == 1) {
#pragma unroll
      for (int nt = 0; nt < 4; ++nt)
#pragma unroll
        for (int r = 0; r < 8; ++r)
          kb[(size_t)(row0 + r + 8 * hf) * DH + nt * 16 + l16] = f2bf(acc[nt][r]);
    } else {
#pragma unroll
      for (int nt = 0; nt < 4; ++nt)
#pragma unroll
        for (int r = 0; r < 8; ++r)
          vtb[(size_t)(nt * 16 + l16) * NTOK + row0 + r + 8 * hf] = f2bf(acc[nt][r]);
    }
  }
}

// ---------------------------------------------------------------------------
// Kernel 2: flash attention with double-buffered adj/k fragment prefetch.
// One wave owns 16 query rows; iterates key blocks of 32. The next block's
// adj C-fragments (HBM stream, non-temporal) and k B-fragments (L2) are
// issued before the current block's WMMA + softmax, so HBM latency hides
// under the exp/shuffle VALU work instead of stalling the loop head.
// ---------------------------------------------------------------------------
struct KFrags { v16bf b00, b01, b10, b11; };

static __device__ inline KFrags ld_kfrags(const unsigned short* __restrict__ kb,
                                          int j0, int l16, int hf) {
  const unsigned short* kr0 = kb + (size_t)(j0 + l16) * DH;
  const unsigned short* kr1 = kb + (size_t)(j0 + 16 + l16) * DH;
  KFrags f;
  f.b00 = ld_b_contig(kr0 + 16 * hf);
  f.b01 = ld_b_contig(kr0 + 32 + 16 * hf);
  f.b10 = ld_b_contig(kr1 + 16 * hf);
  f.b11 = ld_b_contig(kr1 + 32 + 16 * hf);
  return f;
}

static __device__ inline void ld_adjfrags(const float* __restrict__ adjbase,
                                          int j0, v8f& s0, v8f& s1) {
#pragma unroll
  for (int r = 0; r < 8; ++r) {
    s0[r] = __builtin_nontemporal_load(adjbase + (size_t)r * NTOK + j0);
    s1[r] = __builtin_nontemporal_load(adjbase + (size_t)r * NTOK + j0 + 16);
  }
}

__global__ __launch_bounds__(64) void flash_attn(
    const unsigned short* __restrict__ qb,
    const unsigned short* __restrict__ kb,
    const unsigned short* __restrict__ vtb,
    const float* __restrict__ adj,
    float* __restrict__ out)
{
  __shared__ unsigned short pbuf[2][16 * 32]; // per-wave P tile (16x32 bf16)
  const int tid = threadIdx.x;
  const int wv  = tid >> 5;
  const int ln  = tid & 31;
  const int hf  = ln >> 4;
  const int l16 = ln & 15;
  const int row0 = blockIdx.x * 32 + wv * 16;

  // q A-fragments, resident for the whole pass (d = 0..31, 32..63)
  const unsigned short* qrow = qb + (size_t)(row0 + l16) * DH;
  const v16bf Aq0 = ld_a_bf16(qrow, 0, hf);
  const v16bf Aq1 = ld_a_bf16(qrow, 32, hf);

  v8f O[4];
#pragma unroll
  for (int nt = 0; nt < 4; ++nt)
#pragma unroll
    for (int r = 0; r < 8; ++r) O[nt][r] = 0.0f;

  float mrow[8], lrow[8]; // stats for row r + 8*hf, replicated over 16 lanes
#pragma unroll
  for (int r = 0; r < 8; ++r) { mrow[r] = -INFINITY; lrow[r] = 0.0f; }

  // adj element for (r, frag f, block j0): adjbase + r*NTOK + f*16 + j0
  const float* adjbase = adj + (size_t)(row0 + 8 * hf) * NTOK + l16;

  // ---- prologue: prefetch block 0 ----
  v8f s0n, s1n;
  ld_adjfrags(adjbase, 0, s0n, s1n);
  KFrags kn = ld_kfrags(kb, 0, l16, hf);

  for (int j0 = 0; j0 < NTOK; j0 += 32) {
    // take prefetched fragments for this block
    v8f s0 = s0n, s1 = s1n;
    KFrags kc = kn;
    // issue next block's adj (HBM, non-temporal) + k (L2) loads NOW
    const int jn = j0 + 32;
    if (jn < NTOK) {
      ld_adjfrags(adjbase, jn, s0n, s1n);
      kn = ld_kfrags(kb, jn, l16, hf);
    }

    // ---- S = q.k^T * scale + adj (scale pre-folded into q) ----
    s0 = wmma_bf16(Aq0, kc.b00, s0);
    s0 = wmma_bf16(Aq1, kc.b01, s0);
    s1 = wmma_bf16(Aq0, kc.b10, s1);
    s1 = wmma_bf16(Aq1, kc.b11, s1);

    // ---- online softmax over this 32-wide block ----
#pragma unroll
    for (int r = 0; r < 8; ++r) {
      float t = fmaxf(s0[r], s1[r]);
#pragma unroll
      for (int sm = 8; sm >= 1; sm >>= 1) t = fmaxf(t, __shfl_xor(t, sm, 16));
      float mn = fmaxf(mrow[r], t);
      float al = __expf(mrow[r] - mn);
      mrow[r] = mn;
      float p0 = __expf(s0[r] - mn);
      float p1 = __expf(s1[r] - mn);
      float rs = p0 + p1;
#pragma unroll
      for (int sm = 8; sm >= 1; sm >>= 1) rs += __shfl_xor(rs, sm, 16);
      lrow[r] = lrow[r] * al + rs;
#pragma unroll
      for (int nt = 0; nt < 4; ++nt) O[nt][r] *= al;
      // P tile to LDS (C-layout -> memory): row r+8*hf, cols l16 / l16+16
      unsigned short* pp = &pbuf[wv][(r + 8 * hf) * 32 + l16];
      pp[0]  = f2bf(p0);
      pp[16] = f2bf(p1);
    }

    // ---- O += P @ V : reload P as A-fragment, vT rows give B-fragments ----
    v16bf Ap = ld_a_bf16(&pbuf[wv][l16 * 32], 0, hf);
#pragma unroll
    for (int nt = 0; nt < 4; ++nt) {
      v16bf Bv = ld_b_contig(vtb + (size_t)(nt * 16 + l16) * NTOK + j0 + 16 * hf);
      O[nt] = wmma_bf16(Ap, Bv, O[nt]);
    }
  }

  // ---- finalize: divide by softmax denominator, store f32 output ----
#pragma unroll
  for (int r = 0; r < 8; ++r) {
    float inv = 1.0f / lrow[r];
#pragma unroll
    for (int nt = 0; nt < 4; ++nt)
      out[(size_t)(row0 + r + 8 * hf) * DH + nt * 16 + l16] = O[nt][r] * inv;
  }
}

extern "C" void kernel_launch(void* const* d_in, const int* in_sizes, int n_in,
                              void* d_out, int out_size, void* d_ws, size_t ws_size,
                              hipStream_t stream) {
  const float* x   = (const float*)d_in[0];
  const float* adj = (const float*)d_in[1];
  const float* Wq  = (const float*)d_in[2];
  const float* bq_ = (const float*)d_in[3];
  const float* Wk  = (const float*)d_in[4];
  const float* bk_ = (const float*)d_in[5];
  const float* Wv  = (const float*)d_in[6];
  const float* bv_ = (const float*)d_in[7];
  float* out = (float*)d_out;

  unsigned short* qb  = (unsigned short*)d_ws;         // [NTOK][64] bf16 (scaled)
  unsigned short* kb  = qb + (size_t)NTOK * DH;        // [NTOK][64] bf16
  unsigned short* vtb = kb + (size_t)NTOK * DH;        // [64][NTOK] bf16 (v^T)

  proj_qkv<<<NTOK / 128, 256, 0, stream>>>(x, Wq, bq_, Wk, bk_, Wv, bv_,
                                           qb, kb, vtb);
  flash_attn<<<NTOK / 32, 64, 0, stream>>>(qb, kb, vtb, adj, out);
}
